// VectorQuantizer_65429531787925
// MI455X (gfx1250) — matrix operand order repaired and checked by reading, not compiled, not run on hardware
//
#include <hip/hip_runtime.h>

// CDNA5 / gfx1250 VQ codebook lookup.
// d = ||z||^2 + ||c||^2 - 2 z.cT ; argmin over 1024 entries; gather + MSE loss.
// GEMM 65536x1024x256 in f16 WMMA (f32 accum): ~34.4 GFLOP; memory floor ~6us
// at 23.3 TB/s -> f16 WMMA keeps this memory-bound.
// v2: (a) all 16 ds_load_b128 of a sub-tile hoisted ahead of the 8-WMMA chain
//     so waits become partial dscnt counts instead of dscnt==0 per step;
//     (b) codebook tile t+1 global-loaded into registers during compute of
//     tile t (register double-buffer) -> staging is cvt+ds_store only.

typedef _Float16 v16h __attribute__((ext_vector_type(16)));
typedef _Float16 v8h  __attribute__((ext_vector_type(8)));
typedef float    v8f  __attribute__((ext_vector_type(8)));
typedef float    v4f  __attribute__((ext_vector_type(4)));

#define N_E      1024
#define E_DIM    256
#define NPIX     65536          // 16 * 64 * 64
#define M_BLK    128            // rows per workgroup (8 waves x 16)
#define E_TILE   64             // codebook entries staged in LDS per iter
#define CB_STRIDE 264           // 256 halves + 8 pad halves (bank-conflict-free)
#define ZQ_ELEMS 16777216       // 16*256*64*64
#define LOSS_OFF ZQ_ELEMS
#define PERP_OFF (ZQ_ELEMS + 1)
#define IDX_OFF  (ZQ_ELEMS + 2)
#define NUM_BLOCKS (NPIX / M_BLK)   // 512
#define N_TILES  (N_E / E_TILE)     // 16

// ---------- kernel 1: codebook squared norms (f32 exact) ----------
__global__ void vq_cnorm(const float* __restrict__ cb, float* __restrict__ cnorm) {
    int e = blockIdx.x * blockDim.x + threadIdx.x;
    if (e >= N_E) return;
    const v4f* p = (const v4f*)(cb + (size_t)e * E_DIM);
    float s = 0.f;
#pragma unroll
    for (int i = 0; i < E_DIM / 4; ++i) {
        v4f q = p[i];
        s += q[0]*q[0] + q[1]*q[1] + q[2]*q[2] + q[3]*q[3];
    }
    cnorm[e] = s;
}

// ---------- kernel 2: main distance GEMM + argmin + gather ----------
__global__ __launch_bounds__(256) void vq_main(
    const float* __restrict__ z, const float* __restrict__ cb,
    const float* __restrict__ cnorm, float* __restrict__ out,
    float* __restrict__ loss_part)
{
    __shared__ _Float16 lds_cb[E_TILE * CB_STRIDE];   // f16 codebook tile
    __shared__ float    lds_cn[E_TILE];
    __shared__ int      lds_idx[M_BLK];
    __shared__ float    lds_min[M_BLK];
    __shared__ float    lds_zn[M_BLK];
    __shared__ float    lds_red[M_BLK];

    const int tid  = threadIdx.x;
    const int lane = tid & 31;
    const int wv   = tid >> 5;          // wave 0..7
    const int hlf  = lane >> 4;         // lane half (WMMA layout)
    const int l16  = lane & 15;

    const int n0   = blockIdx.x * M_BLK;
    const int nrow = n0 + wv * 16 + l16;        // this lane's query row
    const int bimg = nrow >> 12;                // 4096 = 64*64 pixels / image
    const int hw   = nrow & 4095;
    const float* zrow = z + ((size_t)bimg << 20) + hw;   // C*HW = 1<<20

    // staging geometry (all 256 threads): entry = tid/4, quarter row = tid%4
    const int st_el = tid >> 2;
    const int st_kq = (tid & 3) * 64;
    _Float16* st_dst = &lds_cb[st_el * CB_STRIDE + st_kq];

    // ---- A fragments: 16x32 f16 per k-step, 8 steps (K=256), kept in VGPRs ----
    // element j of v16h maps to K = 16*hlf + j
    v16h a[8];
    float zn = 0.f;
#pragma unroll
    for (int s = 0; s < 8; ++s) {
#pragma unroll
        for (int j = 0; j < 16; ++j) {
            int k = 32 * s + 16 * hlf + j;
            float v = zrow[(size_t)k << 12];    // stride H*W between channels
            zn += v * v;
            a[s][j] = (_Float16)v;
        }
    }
    zn += __shfl_xor(zn, 16, 32);   // full ||z_row||^2 (lane & lane^16 pair)

    float minv[8];
    int   mini[8];
#pragma unroll
    for (int r = 0; r < 8; ++r) { minv[r] = 3.4e38f; mini[r] = 0; }

    // register double-buffer for the codebook tile (64 f32 / thread)
    v4f  creg[16];
    float cnreg = 0.f;

    // prime: load tile 0 into registers
    {
        const v4f* src = (const v4f*)(cb + (size_t)st_el * E_DIM + st_kq);
#pragma unroll
        for (int i = 0; i < 16; ++i) creg[i] = src[i];
        if (tid < E_TILE) cnreg = cnorm[tid];
    }

    for (int t = 0; t < N_TILES; ++t) {
        const int e0 = t * E_TILE;

        // ---- stage registers -> LDS (cvt f32->f16, ds_store_b128) ----
#pragma unroll
        for (int i = 0; i < 8; ++i) {
            v4f q0 = creg[2 * i], q1 = creg[2 * i + 1];
            v8h h;
            h[0] = (_Float16)q0[0]; h[1] = (_Float16)q0[1];
            h[2] = (_Float16)q0[2]; h[3] = (_Float16)q0[3];
            h[4] = (_Float16)q1[0]; h[5] = (_Float16)q1[1];
            h[6] = (_Float16)q1[2]; h[7] = (_Float16)q1[3];
            *(v8h*)(st_dst + 8 * i) = h;
        }
        if (tid < E_TILE) lds_cn[tid] = cnreg;
        __syncthreads();   // LDS tile visible to all waves

        // ---- issue global loads for tile t+1 (latency hidden by compute) ----
        if (t + 1 < N_TILES) {
            const v4f* src = (const v4f*)(cb + (size_t)(e0 + E_TILE + st_el) * E_DIM
                                          + st_kq);
#pragma unroll
            for (int i = 0; i < 16; ++i) creg[i] = src[i];
            if (tid < E_TILE) cnreg = cnorm[e0 + E_TILE + tid];
        }

        // ---- 4 x (16-entry sub-tile): hoisted B loads + 8-WMMA chain ----
#pragma unroll
        for (int es = 0; es < E_TILE / 16; ++es) {
            const _Float16* bbase = &lds_cb[(es * 16 + l16) * CB_STRIDE + hlf * 16];
            v8h bl[8], bh[8];
#pragma unroll
            for (int s = 0; s < 8; ++s) {       // 16x ds_load_b128 up front
                const v8h* bp = (const v8h*)(bbase + s * 32);
                bl[s] = bp[0];
                bh[s] = bp[1];
            }
            v8f acc = {};
#pragma unroll
            for (int s = 0; s < 8; ++s) {
                v16h bf = __builtin_shufflevector(bl[s], bh[s],
                          0,1,2,3,4,5,6,7,8,9,10,11,12,13,14,15);
                acc = __builtin_amdgcn_wmma_f32_16x16x32_f16(
                          false, a[s], false, bf, (short)0, acc, false, false);
            }
            float cn = lds_cn[es * 16 + l16];
            int   eg = e0 + es * 16 + l16;
#pragma unroll
            for (int r = 0; r < 8; ++r) {
                float d = __builtin_fmaf(-2.f, acc[r], cn);  // ||c||^2 - 2 z.c
                if (d < minv[r]) { minv[r] = d; mini[r] = eg; }
            }
        }
        __syncthreads();   // compute done before next LDS overwrite
    }

    // ---- argmin butterfly across the 16 columns of each half ----
#pragma unroll
    for (int m = 1; m < 16; m <<= 1) {
#pragma unroll
        for (int r = 0; r < 8; ++r) {
            float ov = __shfl_xor(minv[r], m, 32);
            int   oi = __shfl_xor(mini[r], m, 32);
            if (ov < minv[r] || (ov == minv[r] && oi < mini[r])) {
                minv[r] = ov; mini[r] = oi;
            }
        }
    }
    // C-fragment slot r on half h holds row (r + 8*h) of the wave's 16 rows
    if (l16 == 0) {
#pragma unroll
        for (int r = 0; r < 8; ++r) {
            lds_idx[wv * 16 + hlf * 8 + r] = mini[r];
            lds_min[wv * 16 + hlf * 8 + r] = minv[r];
        }
    }
    if (hlf == 0) lds_zn[wv * 16 + l16] = zn;
    __syncthreads();

    // ---- indices (as float) + deterministic block loss partial ----
    if (tid < M_BLK) {
        out[IDX_OFF + n0 + tid] = (float)lds_idx[tid];
        lds_red[tid] = lds_min[tid] + lds_zn[tid];   // ||c* - z||^2 per row
    }
    __syncthreads();
    if (tid == 0) {
        float s = 0.f;
        for (int i = 0; i < M_BLK; ++i) s += lds_red[i];
        loss_part[blockIdx.x] = s;
    }

    // ---- z_q gather: out[b][c][hw] = codebook[idx[n]][c], coalesced over rows ----
    {
        int row = tid & 127;
        int cq  = (tid >> 7) * 128;
        int n   = n0 + row;
        int bi  = n >> 12;
        int hwi = n & 4095;
        const v4f* crow = (const v4f*)(cb + (size_t)lds_idx[row] * E_DIM + cq);
        float* orow = out + ((size_t)bi << 20) + hwi + ((size_t)cq << 12);
#pragma unroll
        for (int i = 0; i < 32; ++i) {
            v4f q = crow[i];
            orow[(size_t)(4 * i + 0) << 12] = q[0];
            orow[(size_t)(4 * i + 1) << 12] = q[1];
            orow[(size_t)(4 * i + 2) << 12] = q[2];
            orow[(size_t)(4 * i + 3) << 12] = q[3];
        }
    }
}

// ---------- kernel 3: finalize loss + zero scalar ----------
__global__ void vq_finalize(const float* __restrict__ loss_part,
                            float* __restrict__ out) {
    __shared__ float red[256];
    int t = threadIdx.x;
    red[t] = loss_part[t] + loss_part[t + 256];
    __syncthreads();
    if (t == 0) {
        float s = 0.f;
        for (int i = 0; i < 256; ++i) s += red[i];
        out[LOSS_OFF] = s / (float)ZQ_ELEMS;   // mean over N*C elements
        out[PERP_OFF] = 0.f;
    }
}

extern "C" void kernel_launch(void* const* d_in, const int* in_sizes, int n_in,
                              void* d_out, int out_size, void* d_ws, size_t ws_size,
                              hipStream_t stream) {
    const float* z  = (const float*)d_in[0];
    const float* cb = (const float*)d_in[1];
    float* out       = (float*)d_out;
    float* cnorm     = (float*)d_ws;          // 1024 floats
    float* loss_part = cnorm + N_E;           // 512 floats

    vq_cnorm   <<<4, 256, 0, stream>>>(cb, cnorm);
    vq_main    <<<NUM_BLOCKS, 256, 0, stream>>>(z, cb, cnorm, out, loss_part);
    vq_finalize<<<1, 256, 0, stream>>>(loss_part, out);
}
